// MultiHeadAttention_84954453115654
// MI455X (gfx1250) — matrix operand order repaired and checked by reading, not compiled, not run on hardware
//
#include <hip/hip_runtime.h>
#include <hip/hip_bf16.h>

// MI455X / gfx1250: wave32, WMMA bf16 16x16x32 (fp32 accum) for all GEMM stages.
// Operand layouts chosen so each WMMA fragment is 2 contiguous 16B loads per
// lane (global_load_b128 / ds_load_b128). Working set (~54MB) lives in the
// 192MB L2, so direct global loads with L2 reuse replace LDS staging.
// GEMM uses a 32x64 per-wave tile: each B fragment feeds 2 WMMAs and each A
// fragment 4, doubling wmma density per byte loaded.

typedef __attribute__((ext_vector_type(16))) __bf16 v16bf;
typedef __attribute__((ext_vector_type(8)))  float  v8f;

union AFrag {
    v16bf v;
    unsigned short h[16];
    unsigned int   u[8];
    uint4          q[2];
};

__device__ __forceinline__ v8f wmma_bf16(const AFrag& a, const AFrag& b, v8f c) {
    return __builtin_amdgcn_wmma_f32_16x16x32_bf16(false, a.v, false, b.v,
                                                   (short)0, c, false, false);
}

static constexpr int Bsz = 2, EMBED = 512, HEADS = 8, T = 2048, DH = 64;
static constexpr float EPS = 1e-5f;
static constexpr float INV_SCALE = 1.0f / 181.0f;   // 512 // sqrt(8) == 181.0

// ---- workspace layout (bytes), total ~54 MB ----
static constexpr size_t SZ_W    = (size_t)EMBED * EMBED * 2;       // bf16 weights
static constexpr size_t SZ_X    = (size_t)Bsz * EMBED * T * 2;     // bf16 activations
static constexpr size_t SZ_P    = (size_t)Bsz * EMBED * T * 4;     // f32 projections
static constexpr size_t OFF_WQN = 0;
static constexpr size_t OFF_WKN = OFF_WQN + SZ_W;
static constexpr size_t OFF_WVN = OFF_WKN + SZ_W;
static constexpr size_t OFF_WON = OFF_WVN + SZ_W;
static constexpr size_t OFF_XQ  = OFF_WON + SZ_W;   // bf16 [B,T,E]  (K-major for GEMM B)
static constexpr size_t OFF_XK  = OFF_XQ + SZ_X;
static constexpr size_t OFF_XV  = OFF_XK + SZ_X;
static constexpr size_t OFF_PQ  = OFF_XV + SZ_X;    // f32 [B,E,T]
static constexpr size_t OFF_PK  = OFF_PQ + SZ_P;
static constexpr size_t OFF_PV  = OFF_PK + SZ_P;
static constexpr size_t OFF_QH  = OFF_PV + SZ_P;    // bf16 [B,H,T,DH]
static constexpr size_t OFF_KH  = OFF_QH + SZ_X;    // bf16 [B,H,T,DH]
static constexpr size_t OFF_VT  = OFF_KH + SZ_X;    // bf16 [B,H,DH,T]  (transposed V)
static constexpr size_t OFF_AT  = OFF_VT + SZ_X;    // bf16 [B,T,E]   attention output

// ============================================================================
// Stage 1: weight standardization (one wave per 512-wide row) -> bf16
// ============================================================================
__global__ __launch_bounds__(256) void ws_weights_kernel(
    const float* __restrict__ W0, const float* __restrict__ W1,
    const float* __restrict__ W2, const float* __restrict__ W3,
    __bf16* __restrict__ O0, __bf16* __restrict__ O1,
    __bf16* __restrict__ O2, __bf16* __restrict__ O3)
{
    const float* W; __bf16* O;
    switch (blockIdx.y) {
        case 0:  W = W0; O = O0; break;
        case 1:  W = W1; O = O1; break;
        case 2:  W = W2; O = O2; break;
        default: W = W3; O = O3; break;
    }
    const int lane = threadIdx.x & 31, wave = threadIdx.x >> 5;
    const int row = blockIdx.x * 8 + wave;                  // < 512
    const float* wr = W + (size_t)row * EMBED;
    float s = 0.f, s2 = 0.f;
    for (int i = lane; i < EMBED; i += 32) { float x = wr[i]; s += x; s2 += x * x; }
    for (int m = 1; m < 32; m <<= 1) { s += __shfl_xor(s, m, 32); s2 += __shfl_xor(s2, m, 32); }
    const float mu  = s * (1.f / EMBED);
    const float var = s2 * (1.f / EMBED) - mu * mu;
    const float rs  = rsqrtf(var + EPS);
    __bf16* orow = O + (size_t)row * EMBED;
    for (int i = lane; i < EMBED; i += 32) orow[i] = (__bf16)((wr[i] - mu) * rs);
}

// ============================================================================
// Stage 2: mask + bf16 convert + transpose to [B,T,E] via LDS tile.
// grid: (T/32, E/32, 3*B); block 256. Coalesced reads and writes.
// ============================================================================
__global__ __launch_bounds__(256) void mask_transpose_kernel(
    const float* __restrict__ q, const float* __restrict__ k, const float* __restrict__ v,
    const int* __restrict__ qm, const int* __restrict__ km, const int* __restrict__ vm,
    __bf16* __restrict__ xq, __bf16* __restrict__ xk, __bf16* __restrict__ xv)
{
    __shared__ float tile[32][33];
    const int tensor = blockIdx.z / Bsz, b = blockIdx.z % Bsz;
    const float* X; const int* M; __bf16* O;
    switch (tensor) {
        case 0:  X = q; M = qm; O = xq; break;
        case 1:  X = k; M = km; O = xk; break;
        default: X = v; M = vm; O = xv; break;
    }
    const int t0 = blockIdx.x * 32, i0 = blockIdx.y * 32;
    const int tx = threadIdx.x & 31, ty = threadIdx.x >> 5;   // ty in [0,8)
    const float mf = (M[b * T + t0 + tx] != 0) ? 1.f : 0.f;
#pragma unroll
    for (int j = 0; j < 4; ++j) {
        const int i = ty + 8 * j;
        tile[i][tx] = X[((size_t)b * EMBED + i0 + i) * T + t0 + tx] * mf;
    }
    __syncthreads();
#pragma unroll
    for (int j = 0; j < 4; ++j) {
        const int t = ty + 8 * j;
        O[((size_t)b * T + t0 + t) * EMBED + i0 + tx] = (__bf16)tile[tx][t];
    }
}

// ============================================================================
// Stage 3/6: bf16 WMMA GEMM, fixed 512x2048x512 per batch.
// C[b](MxN) = A(MxK,row-major) * BT[b](NxK, K-major) + bias.
// 32x64 tile per wave (2 A-frags x 4 B-frags -> 8 WMMA/K-step), 8 waves/block.
// ============================================================================
__global__ __launch_bounds__(256) void gemm_bf16_kernel(
    const __bf16* __restrict__ A, const __bf16* __restrict__ BT,
    const float* __restrict__ bias, float* __restrict__ C)
{
    constexpr int M = EMBED, N = T, K = EMBED;
    const int b = blockIdx.y;
    const __bf16* Bx = BT + (size_t)b * N * K;   // [N][K]
    float*        Cx = C  + (size_t)b * M * N;   // [M][N]
    const int lane = threadIdx.x & 31, wave = threadIdx.x >> 5;
    const int hl = lane >> 4, l15 = lane & 15;
    constexpr int tilesN = N >> 6;               // N/64 tiles in N
    const int tile = blockIdx.x * 8 + wave;      // over (M/32)*(N/64)
    const int tm = tile / tilesN, tn = tile % tilesN;

    const __bf16* Arow[2];
#pragma unroll
    for (int i = 0; i < 2; ++i)
        Arow[i] = A + (size_t)(tm * 32 + i * 16 + l15) * K;
    const __bf16* Brow[4];
#pragma unroll
    for (int j = 0; j < 4; ++j)
        Brow[j] = Bx + (size_t)(tn * 64 + j * 16 + l15) * K + 16 * hl;

    v8f acc[2][4];
#pragma unroll
    for (int i = 0; i < 2; ++i)
#pragma unroll
        for (int j = 0; j < 4; ++j) acc[i][j] = (v8f){};

    for (int kc = 0; kc < K; kc += 32) {
        AFrag aF[2], bF[4];
#pragma unroll
        for (int i = 0; i < 2; ++i) {
            aF[i].q[0] = *(const uint4*)(Arow[i] + kc + 16 * hl);     // e 0..7 : K=16*hl+e
            aF[i].q[1] = *(const uint4*)(Arow[i] + kc + 16 * hl + 8); // e 8..15: K=16*hl+e
        }
#pragma unroll
        for (int j = 0; j < 4; ++j) {
            bF[j].q[0] = *(const uint4*)(Brow[j] + kc);               // e: K=16*hl+e
            bF[j].q[1] = *(const uint4*)(Brow[j] + kc + 8);
        }
#pragma unroll
        for (int i = 0; i < 2; ++i)
#pragma unroll
            for (int j = 0; j < 4; ++j)
                acc[i][j] = wmma_bf16(aF[i], bF[j], acc[i][j]);
    }

#pragma unroll
    for (int i = 0; i < 2; ++i)
#pragma unroll
        for (int r = 0; r < 8; ++r) {
            const int row = tm * 32 + i * 16 + r + 8 * hl;
            const float bi = bias[row];
#pragma unroll
            for (int j = 0; j < 4; ++j)
                Cx[(size_t)row * N + tn * 64 + j * 16 + l15] = acc[i][j][r] + bi;
        }
}

// ============================================================================
// Stage 4a: per-head LayerNorm over DH=64 -> bf16 head-major [B,H,T,DH] (Q,K)
// ============================================================================
__global__ __launch_bounds__(256) void ln_head_kernel(
    const float* __restrict__ P, const float* __restrict__ g,
    const float* __restrict__ bb, __bf16* __restrict__ O)
{
    const int lane = threadIdx.x & 31, wave = threadIdx.x >> 5;
    const int gidx = blockIdx.x * 8 + wave;          // < B*H*T
    const int t  = gidx % T;
    const int bh = gidx / T;
    const int h = bh & (HEADS - 1), b = bh / HEADS;
    const float* base = P + ((size_t)b * EMBED + h * DH) * T + t;
    const float x0 = base[(size_t)lane * T];
    const float x1 = base[(size_t)(lane + 32) * T];
    float s = x0 + x1, s2 = x0 * x0 + x1 * x1;
    for (int m = 1; m < 32; m <<= 1) { s += __shfl_xor(s, m, 32); s2 += __shfl_xor(s2, m, 32); }
    const float mu  = s * (1.f / DH);
    const float var = s2 * (1.f / DH) - mu * mu;
    const float rs  = rsqrtf(var + EPS);
    __bf16* o = O + ((size_t)bh * T + t) * DH;
    o[lane]      = (__bf16)((x0 - mu) * rs * g[lane]      + bb[lane]);
    o[lane + 32] = (__bf16)((x1 - mu) * rs * g[lane + 32] + bb[lane + 32]);
}

// ============================================================================
// Stage 4b: same LayerNorm but transposed store -> bf16 [B,H,DH,T] (V)
// ============================================================================
__global__ __launch_bounds__(256) void ln_head_t_kernel(
    const float* __restrict__ P, const float* __restrict__ g,
    const float* __restrict__ bb, __bf16* __restrict__ OT)
{
    const int lane = threadIdx.x & 31, wave = threadIdx.x >> 5;
    const int gidx = blockIdx.x * 8 + wave;          // < B*H*T
    const int t  = gidx % T;
    const int bh = gidx / T;
    const int h = bh & (HEADS - 1), b = bh / HEADS;
    const float* base = P + ((size_t)b * EMBED + h * DH) * T + t;
    const float x0 = base[(size_t)lane * T];
    const float x1 = base[(size_t)(lane + 32) * T];
    float s = x0 + x1, s2 = x0 * x0 + x1 * x1;
    for (int m = 1; m < 32; m <<= 1) { s += __shfl_xor(s, m, 32); s2 += __shfl_xor(s2, m, 32); }
    const float mu  = s * (1.f / DH);
    const float var = s2 * (1.f / DH) - mu * mu;
    const float rs  = rsqrtf(var + EPS);
    __bf16* o = OT + (size_t)bh * DH * T;            // [DH][T]
    o[(size_t)lane * T + t]        = (__bf16)((x0 - mu) * rs * g[lane]      + bb[lane]);
    o[(size_t)(lane + 32) * T + t] = (__bf16)((x1 - mu) * rs * g[lane + 32] + bb[lane + 32]);
}

// ============================================================================
// Stage 5: flash-style attention. Block=128 (4 waves); one 16-row query tile
// per wave; online softmax over 32-key chunks.
//   V fragments hoisted to iteration start (overlap the whole softmax block);
//   score fragments batch-loaded before the 4 score WMMAs.
//   P tile transposed C-layout -> A-layout through per-wave LDS (b128 aligned).
// ============================================================================
__global__ __launch_bounds__(128) void attention_kernel(
    const __bf16* __restrict__ Q, const __bf16* __restrict__ Kh,
    const __bf16* __restrict__ VT, const int* __restrict__ qmask,
    const int* __restrict__ kmask, __bf16* __restrict__ Aout)
{
    __shared__ alignas(16) __bf16 ldsP[4][16][40];   // 80B rows -> 16B-aligned reads

    const int lane = threadIdx.x & 31, wave = threadIdx.x >> 5;
    const int hl = lane >> 4, l15 = lane & 15;
    const int bh = blockIdx.y;                       // < B*H
    const int b = bh >> 3, h = bh & 7;
    const int qtile = blockIdx.x * 4 + wave;         // < T/16
    const int qbase = qtile * 16;

    const __bf16* Qb = Q  + (size_t)bh * T * DH;     // [T][DH]
    const __bf16* Kb = Kh + (size_t)bh * T * DH;     // [T][DH]
    const __bf16* Vb = VT + (size_t)bh * DH * T;     // [DH][T]
    const int* kmb = kmask + b * T;

    // Q A-fragments for dh chunks {0..31} and {32..63} (loaded once per tile).
    AFrag aQ[2];
    const __bf16* qrow = Qb + (size_t)(qbase + l15) * DH;
#pragma unroll
    for (int c = 0; c < 2; ++c) {
        aQ[c].q[0] = *(const uint4*)(qrow + c * 32 + 16 * hl);
        aQ[c].q[1] = *(const uint4*)(qrow + c * 32 + 16 * hl + 8);
    }

    float mrun[8], lrun[8];
    v8f oacc[4];
#pragma unroll
    for (int r = 0; r < 8; ++r) { mrun[r] = -3.0e38f; lrun[r] = 0.f; }
#pragma unroll
    for (int d = 0; d < 4; ++d) oacc[d] = (v8f){};

    for (int kt = 0; kt < T; kt += 32) {
        // ---- hoisted V fragments (consumed only after the softmax block) ----
        AFrag bV[4];
#pragma unroll
        for (int dt = 0; dt < 4; ++dt) {
            const __bf16* vrow = Vb + (size_t)(dt * 16 + l15) * T + kt + 16 * hl;
            bV[dt].q[0] = *(const uint4*)(vrow);           // kk = 16*hl + e
            bV[dt].q[1] = *(const uint4*)(vrow + 8);
        }
        // ---- key mask + both key subtiles' fragments ----
        int kmv[2];
        kmv[0] = kmb[kt + l15];
        kmv[1] = kmb[kt + 16 + l15];
        AFrag bK[2][2];
#pragma unroll
        for (int sub = 0; sub < 2; ++sub) {
            const __bf16* krow = Kb + (size_t)(kt + sub * 16 + l15) * DH + 16 * hl;
            bK[sub][0].q[0] = *(const uint4*)(krow);       // dh = 16*hl + e
            bK[sub][0].q[1] = *(const uint4*)(krow + 8);
            bK[sub][1].q[0] = *(const uint4*)(krow + 32);  // dh = 32 + 16*hl + e
            bK[sub][1].q[1] = *(const uint4*)(krow + 40);
        }
        // ---- scores: 4 WMMAs (two subtiles, K=DH=64) ----
        v8f sfrag[2];
#pragma unroll
        for (int sub = 0; sub < 2; ++sub) {
            v8f s = {};
            s = wmma_bf16(aQ[0], bK[sub][0], s);
            s = wmma_bf16(aQ[1], bK[sub][1], s);
            const bool masked = (kmv[sub] == 0);
#pragma unroll
            for (int r = 0; r < 8; ++r)
                s[r] = masked ? -1e9f : (s[r] * INV_SCALE);
            sfrag[sub] = s;
        }
        // ---- online softmax row stats (rows live in half-waves of 16 lanes)
        float mnew[8], fr[8];
#pragma unroll
        for (int r = 0; r < 8; ++r) {
            float mx = fmaxf(sfrag[0][r], sfrag[1][r]);
            for (int msk = 1; msk < 16; msk <<= 1) mx = fmaxf(mx, __shfl_xor(mx, msk, 32));
            mnew[r] = fmaxf(mrun[r], mx);
            fr[r]   = __expf(mrun[r] - mnew[r]);
            mrun[r] = mnew[r];
        }
#pragma unroll
        for (int d = 0; d < 4; ++d)
#pragma unroll
            for (int r = 0; r < 8; ++r) oacc[d][r] *= fr[r];

        // ---- P = exp(s - m): row sums + LDS stash in (row, key) order ----
        float padd[8];
#pragma unroll
        for (int r = 0; r < 8; ++r) padd[r] = 0.f;
#pragma unroll
        for (int sub = 0; sub < 2; ++sub)
#pragma unroll
            for (int r = 0; r < 8; ++r) {
                const float p = __expf(sfrag[sub][r] - mnew[r]);
                padd[r] += p;
                ldsP[wave][r + 8 * hl][sub * 16 + l15] = (__bf16)p;
            }
#pragma unroll
        for (int r = 0; r < 8; ++r) {
            float ps = padd[r];
            for (int msk = 1; msk < 16; msk <<= 1) ps += __shfl_xor(ps, msk, 32);
            lrun[r] = lrun[r] * fr[r] + ps;
        }
        asm volatile("s_wait_dscnt 0" ::: "memory");

        // ---- read P back as A-fragment (16x32), b128-aligned ----
        AFrag aP;
        aP.q[0] = *(const uint4*)(&ldsP[wave][l15][16 * hl]);
        aP.q[1] = *(const uint4*)(&ldsP[wave][l15][16 * hl + 8]);

        // ---- O += P * V over four 16-wide d-tiles (V preloaded) ----
#pragma unroll
        for (int dt = 0; dt < 4; ++dt)
            oacc[dt] = wmma_bf16(aP, bV[dt], oacc[dt]);
    }

    // ---- epilogue: 1/l, query-mask zeroing, store bf16 [B,T,E] (coalesced) --
#pragma unroll
    for (int r = 0; r < 8; ++r) {
        const int row = qbase + r + 8 * hl;
        const float qv = (qmask[b * T + row] != 0) ? 1.f : 0.f;
        const float inv = qv / lrun[r];
#pragma unroll
        for (int dt = 0; dt < 4; ++dt) {
            const int d = dt * 16 + l15;
            Aout[((size_t)b * T + row) * EMBED + h * DH + d] = (__bf16)(oacc[dt][r] * inv);
        }
    }
}

// ============================================================================
extern "C" void kernel_launch(void* const* d_in, const int* in_sizes, int n_in,
                              void* d_out, int out_size, void* d_ws, size_t ws_size,
                              hipStream_t stream) {
    const float* q  = (const float*)d_in[0];
    const float* k  = (const float*)d_in[1];
    const float* v  = (const float*)d_in[2];
    const int* qm   = (const int*)d_in[3];
    const int* km   = (const int*)d_in[4];
    const int* vm   = (const int*)d_in[5];
    const float* Wq = (const float*)d_in[6];  const float* bq = (const float*)d_in[7];
    const float* Wk = (const float*)d_in[8];  const float* bk = (const float*)d_in[9];
    const float* Wv = (const float*)d_in[10]; const float* bv = (const float*)d_in[11];
    const float* Wo = (const float*)d_in[12]; const float* bo = (const float*)d_in[13];
    const float* gq = (const float*)d_in[14]; const float* bgq = (const float*)d_in[15];
    const float* gk = (const float*)d_in[16]; const float* bgk = (const float*)d_in[17];
    const float* gv = (const float*)d_in[18]; const float* bgv = (const float*)d_in[19];
    float* out = (float*)d_out;

    char* ws = (char*)d_ws;
    __bf16* wqn = (__bf16*)(ws + OFF_WQN);
    __bf16* wkn = (__bf16*)(ws + OFF_WKN);
    __bf16* wvn = (__bf16*)(ws + OFF_WVN);
    __bf16* won = (__bf16*)(ws + OFF_WON);
    __bf16* xq  = (__bf16*)(ws + OFF_XQ);
    __bf16* xk  = (__bf16*)(ws + OFF_XK);
    __bf16* xv  = (__bf16*)(ws + OFF_XV);
    float*  pq  = (float*)(ws + OFF_PQ);
    float*  pk  = (float*)(ws + OFF_PK);
    float*  pv  = (float*)(ws + OFF_PV);
    __bf16* qh  = (__bf16*)(ws + OFF_QH);
    __bf16* kh  = (__bf16*)(ws + OFF_KH);
    __bf16* vt  = (__bf16*)(ws + OFF_VT);
    __bf16* at  = (__bf16*)(ws + OFF_AT);

    // 1) weight standardization (4 matrices)
    ws_weights_kernel<<<dim3(64, 4), 256, 0, stream>>>(Wq, Wk, Wv, Wo, wqn, wkn, wvn, won);
    // 2) mask + bf16 + transpose to [B,T,E]
    mask_transpose_kernel<<<dim3(T / 32, EMBED / 32, 3 * Bsz), 256, 0, stream>>>(
        q, k, v, qm, km, vm, xq, xk, xv);
    // 3) projections: [512x512] x [512x2048] per batch; (M/32)*(N/64)/8 = 64 blocks
    gemm_bf16_kernel<<<dim3(64, Bsz), 256, 0, stream>>>(wqn, xq, bq, pq);
    gemm_bf16_kernel<<<dim3(64, Bsz), 256, 0, stream>>>(wkn, xk, bk, pk);
    gemm_bf16_kernel<<<dim3(64, Bsz), 256, 0, stream>>>(wvn, xv, bv, pv);
    // 4) per-head LayerNorm: Q,K head-major; V transposed [B,H,DH,T]
    ln_head_kernel  <<<(Bsz * HEADS * T) / 8, 256, 0, stream>>>(pq, gq, bgq, qh);
    ln_head_kernel  <<<(Bsz * HEADS * T) / 8, 256, 0, stream>>>(pk, gk, bgk, kh);
    ln_head_t_kernel<<<(Bsz * HEADS * T) / 8, 256, 0, stream>>>(pv, gv, bgv, vt);
    // 5) flash attention -> [B,T,E]
    attention_kernel<<<dim3((T / 16) / 4, Bsz * HEADS), 128, 0, stream>>>(
        qh, kh, vt, qm, km, at);
    // 6) output projection -> d_out (fp32 [B,E,T])
    gemm_bf16_kernel<<<dim3(64, Bsz), 256, 0, stream>>>(won, at, bo, out);
    (void)in_sizes; (void)n_in; (void)out_size; (void)ws_size;
}
